// DecompModel_35270271435192
// MI455X (gfx1250) — compile-verified, matrix operand order checked by reading, not run on hardware
//
#include <hip/hip_runtime.h>
#include <hip/hip_bf16.h>

// ---------------- types for WMMA ----------------
typedef __bf16 bf16_t;
typedef bf16_t v16bf __attribute__((ext_vector_type(16)));
typedef float  v8f   __attribute__((ext_vector_type(8)));

union FragBF { v16bf v; unsigned u[8]; unsigned short h[16]; };
union FragF  { v8f   v; float f[8]; };

__device__ __forceinline__ unsigned short f2bf(float x) {
    unsigned u = __float_as_uint(x);
    unsigned r = u + 0x7FFFu + ((u >> 16) & 1u);   // round to nearest even
    return (unsigned short)(r >> 16);
}

// gfx1250 async global->LDS copy (16B per lane), tracked by ASYNCcnt
__device__ __forceinline__ void async_copy16(void* ldsPtr, const void* gptr) {
    unsigned ldsOff = (unsigned)(size_t)ldsPtr;            // LDS aperture low bits
    asm volatile("global_load_async_to_lds_b128 %0, %1, off"
                 :: "v"(ldsOff), "v"((unsigned long long)(size_t)gptr)
                 : "memory");
}
__device__ __forceinline__ void async_wait() {
    asm volatile("s_wait_asynccnt 0" ::: "memory");
}

// ---------------- GEMM flags (epilogue only) ----------------
#define F_RELU    1
#define F_SKIPF32 2
#define F_BF16OUT 4

// 16x16x32 bf16 WMMA step: one A fragment, two B fragments (two N sub-tiles)
__device__ __forceinline__ void wmma_step(const unsigned short (*As)[40],
                                          const unsigned short (*Bs)[40],
                                          int wm, int wn, int half, int mr,
                                          v8f& acc0, v8f& acc1)
{
    const unsigned* arow = (const unsigned*)(&As[wm * 16 + mr][0]);
    const unsigned* b0   = (const unsigned*)(&Bs[wn * 32 + mr][0]);
    const unsigned* b1   = (const unsigned*)(&Bs[wn * 32 + 16 + mr][0]);
    FragBF af, bf0, bf1;
    #pragma unroll
    for (int j = 0; j < 4; ++j) {
        af.u[j]      = arow[half * 4 + j];      // K = 8*half + 0..7
        af.u[4 + j]  = arow[8 + half * 4 + j];  // K = 16 + 8*half + 0..7
        bf0.u[j]     = b0[half * 4 + j];
        bf0.u[4 + j] = b0[8 + half * 4 + j];
        bf1.u[j]     = b1[half * 4 + j];
        bf1.u[4 + j] = b1[8 + half * 4 + j];
    }
    acc0 = __builtin_amdgcn_wmma_f32_16x16x32_bf16(false, af.v, false, bf0.v,
                                                   (short)0, acc0, false, false);
    acc1 = __builtin_amdgcn_wmma_f32_16x16x32_bf16(false, af.v, false, bf1.v,
                                                   (short)0, acc1, false, false);
}

// Generic batched WMMA GEMM: C[z] = act(A[z] @ B[z] + bias[z] + R[z])
// A: M x K bf16. B: K x N (f32 or bf16), optionally stored transposed (N x K).
// Block tile 64x64, K step 32, 8 waves, each wave -> 16x32 of C.
template <int BBF, int BTR>
__global__ void k_gemm(const unsigned short* A, long lda, long abst,
                       const void* B_, long ldb, long bbst,
                       float* C, long ldc, long cbst,
                       unsigned short* Cb, long cbbst,
                       const float* bias, long biasbst,
                       const float* resid, long rbst,
                       int M, int N, int K, int flags)
{
    __shared__ unsigned short As[64][40];   // [m][k]
    __shared__ unsigned short Bs[64][40];   // [n][k] (transposed for contiguous frags)

    const int tid  = threadIdx.x;
    const int wid  = tid >> 5;
    const int lane = tid & 31;
    const int half = lane >> 4;
    const int mr   = lane & 15;
    const int wm   = wid >> 1;      // 0..3 (M sub-tile)
    const int wn   = wid & 1;       // 0..1 (N sub-tile pair)
    const int bz   = blockIdx.z;

    const int mBase = blockIdx.y * 64;
    const int nBase = blockIdx.x * 64;
    const int m0 = mBase + wm * 16;
    const int n0 = nBase + wn * 32;

    const unsigned short* Bb = (const unsigned short*)B_;
    const float*          Bf = (const float*)B_;

    v8f acc0 = {}, acc1 = {};

    const bool full = (mBase + 64 <= M) && (nBase + 64 <= N) && ((K & 31) == 0);

    if (full) {
        // ---------- branch-free fast path ----------
        const int ar  = tid >> 2;          // A row 0..63
        const int ac0 = (tid & 3) * 8;     // A k base
        const int gmA = mBase + ar;
        for (int k0 = 0; k0 < K; k0 += 32) {
            // A: bf16, 16B per thread, async global->LDS
            async_copy16(&As[ar][ac0],
                         &A[(long)bz * abst + (long)gmA * lda + (k0 + ac0)]);
            if constexpr (BBF && BTR) {
                // B stored N x K, bf16: async 16B copies along K
                int bn  = tid >> 2;            // 0..63
                int bk0 = (tid & 3) * 8;
                async_copy16(&Bs[bn][bk0],
                             &Bb[(long)bz * bbst + (long)(nBase + bn) * ldb + (k0 + bk0)]);
            } else if constexpr (BBF) {
                // B stored K x N, bf16: vector load along N, scatter-transpose to LDS
                int bk  = tid >> 3;            // 0..31
                int bn0 = (tid & 7) * 8;
                uint4 t = *(const uint4*)&Bb[(long)bz * bbst + (long)(k0 + bk) * ldb + (nBase + bn0)];
                const unsigned short* ts = (const unsigned short*)&t;
                #pragma unroll
                for (int j = 0; j < 8; ++j) Bs[bn0 + j][bk] = ts[j];
            } else {
                // B stored K x N, f32: scalar loads (ldb may be odd), convert+transpose
                int bk  = tid >> 3;            // 0..31
                int bn0 = (tid & 7) * 8;
                long base = (long)bz * bbst + (long)(k0 + bk) * ldb + (nBase + bn0);
                if (k0 + 32 < K)
                    __builtin_prefetch(Bf + base + 32 * ldb, 0, 1);
                #pragma unroll
                for (int j = 0; j < 8; ++j) Bs[bn0 + j][bk] = f2bf(Bf[base + j]);
            }
            async_wait();
            __syncthreads();
            wmma_step(As, Bs, wm, wn, half, mr, acc0, acc1);
            __syncthreads();
        }
    } else {
        // ---------- guarded safe path (edge tiles) ----------
        const int ar  = tid >> 2;
        const int ac0 = (tid & 3) * 8;
        const int gmA = mBase + ar;
        const int bk  = tid >> 3;
        const int bn0 = (tid & 7) * 8;
        for (int k0 = 0; k0 < K; k0 += 32) {
            #pragma unroll
            for (int j = 0; j < 8; ++j) {
                int gk = k0 + ac0 + j;
                As[ar][ac0 + j] = (gmA < M && gk < K)
                    ? A[(long)bz * abst + (long)gmA * lda + gk] : (unsigned short)0;
            }
            #pragma unroll
            for (int j = 0; j < 8; ++j) {
                int gn = nBase + bn0 + j;
                int gk = k0 + bk;
                unsigned short hv = 0;
                if (gn < N && gk < K) {
                    long idx = (long)bz * bbst +
                               (BTR ? ((long)gn * ldb + gk) : ((long)gk * ldb + gn));
                    hv = BBF ? Bb[idx] : f2bf(Bf[idx]);
                }
                Bs[bn0 + j][bk] = hv;
            }
            __syncthreads();
            wmma_step(As, Bs, wm, wn, half, mr, acc0, acc1);
            __syncthreads();
        }
    }

    // ---- epilogue: C layout lanes0-15 VGPRj->M=j ; lanes16-31 VGPRj->M=8+j ----
    FragF f0, f1; f0.v = acc0; f1.v = acc1;
    #pragma unroll
    for (int j = 0; j < 8; ++j) {
        int row = m0 + half * 8 + j;
        if (row >= M) continue;
        #pragma unroll
        for (int t = 0; t < 2; ++t) {
            int col = n0 + t * 16 + mr;
            if (col >= N) continue;
            float v = t ? f1.f[j] : f0.f[j];
            if (bias)  v += bias[(long)bz * biasbst + col];
            if (resid) v += resid[(long)bz * rbst + (long)row * ldc + col];
            if (flags & F_RELU) v = fmaxf(v, 0.0f);
            long o = (long)row * ldc + col;
            if (!(flags & F_SKIPF32)) C[(long)bz * cbst + o] = v;
            if (flags & F_BF16OUT)    Cb[(long)bz * cbbst + o] = f2bf(v);
        }
    }
}

// ---------------- f32 -> bf16 weight conversion ----------------
__global__ void k_cvt_bf16(const float* in, unsigned short* out, long n)
{
    long i = (long)blockIdx.x * blockDim.x + threadIdx.x;
    if (i < n) out[i] = f2bf(in[i]);
}

// ---------------- embedding gather ----------------
__global__ void k_embed(const int* seq, const float* emb, float* h0,
                        unsigned short* h0b, long total, int D)
{
    long i = (long)blockIdx.x * blockDim.x + threadIdx.x;
    if (i >= total) return;
    long t = i / D; int d = (int)(i % D);
    float v = emb[(long)seq[t] * D + d];
    h0[i] = v; h0b[i] = f2bf(v);
}

// ---------------- LayerNorm (in-place) + bf16 copy ----------------
__global__ void k_ln(float* x, unsigned short* hb, const float* g,
                     const float* b, int D)
{
    int row = blockIdx.x, tid = threadIdx.x;
    float* xr = x + (long)row * D;
    __shared__ float red[256];
    float s = 0.f;
    for (int d = tid; d < D; d += 256) s += xr[d];
    red[tid] = s; __syncthreads();
    for (int o = 128; o; o >>= 1) { if (tid < o) red[tid] += red[tid + o]; __syncthreads(); }
    float mu = red[0] / D; __syncthreads();
    s = 0.f;
    for (int d = tid; d < D; d += 256) { float t = xr[d] - mu; s += t * t; }
    red[tid] = s; __syncthreads();
    for (int o = 128; o; o >>= 1) { if (tid < o) red[tid] += red[tid + o]; __syncthreads(); }
    float rstd = rsqrtf(red[0] / D + 1e-5f);
    for (int d = tid; d < D; d += 256) {
        float v = (xr[d] - mu) * rstd * g[d] + b[d];
        xr[d] = v; hb[(long)row * D + d] = f2bf(v);
    }
}

// ---------------- wave-per-row dot with scalar bias (+optional sigmoid) ----------------
__global__ void k_rowdot(const float* X, long ld, long bst, int rowsPerB, int nB,
                         const float* w, const float* bsc, int D, int sigm,
                         float* out, long obst)
{
    int gw   = (int)(((long)blockIdx.x * blockDim.x + threadIdx.x) >> 5);
    int lane = threadIdx.x & 31;
    if (gw >= nB * rowsPerB) return;
    int b = gw / rowsPerB, r = gw % rowsPerB;
    const float* xr = X + (long)b * bst + (long)r * ld;
    float s = 0.f;
    for (int d = lane; d < D; d += 32) s += xr[d] * w[d];
    for (int o = 16; o; o >>= 1) s += __shfl_down(s, o, 32);
    if (!lane) {
        s += bsc[0];
        if (sigm) s = 1.f / (1.f + expf(-s));
        out[(long)b * obst + r] = s;
    }
}

// ---------------- top-k (forward): iterative argmax, then sort ascending ----------------
__global__ void k_topk_fwd(const float* scores, int* idx_out, int nC, int KSEL)
{
    int b = blockIdx.x, tid = threadIdx.x;
    __shared__ float sv[512];
    __shared__ float rv[256]; __shared__ int ri[256];
    __shared__ int picked[64];
    for (int i = tid; i < nC; i += 256) sv[i] = scores[(long)b * nC + i];
    __syncthreads();
    for (int s = 0; s < KSEL; ++s) {
        float best = -1e30f; int bi = nC;
        for (int i = tid; i < nC; i += 256) {
            float v = sv[i];
            if (v > best || (v == best && i < bi)) { best = v; bi = i; }
        }
        rv[tid] = best; ri[tid] = bi; __syncthreads();
        for (int o = 128; o; o >>= 1) {
            if (tid < o && (rv[tid + o] > rv[tid] ||
                (rv[tid + o] == rv[tid] && ri[tid + o] < ri[tid]))) {
                rv[tid] = rv[tid + o]; ri[tid] = ri[tid + o];
            }
            __syncthreads();
        }
        if (!tid) { picked[s] = ri[0]; sv[ri[0]] = -1e30f; }
        __syncthreads();
    }
    if (!tid) {
        for (int i = 1; i < KSEL; ++i) {
            int v = picked[i], j = i - 1;
            while (j >= 0 && picked[j] > v) { picked[j + 1] = picked[j]; --j; }
            picked[j + 1] = v;
        }
        for (int i = 0; i < KSEL; ++i) idx_out[b * KSEL + i] = picked[i];
    }
}

// ---------------- top-k (retro): mask fwd picks, keep descending pick order ----------------
__global__ void k_topk_retro(const float* gs, const int* fidx, int* sel,
                             int nC, int nFwd, int KSEL)
{
    int b = blockIdx.x, tid = threadIdx.x;
    __shared__ float sv[512];
    __shared__ float rv[256]; __shared__ int ri[256];
    for (int i = tid; i < nC; i += 256) sv[i] = gs[(long)b * nC + i];
    __syncthreads();
    if (tid < nFwd) sv[fidx[b * nFwd + tid]] = -1e9f;
    __syncthreads();
    for (int s = 0; s < KSEL; ++s) {
        float best = -1e30f; int bi = nC;
        for (int i = tid; i < nC; i += 256) {
            float v = sv[i];
            if (v > best || (v == best && i < bi)) { best = v; bi = i; }
        }
        rv[tid] = best; ri[tid] = bi; __syncthreads();
        for (int o = 128; o; o >>= 1) {
            if (tid < o && (rv[tid + o] > rv[tid] ||
                (rv[tid + o] == rv[tid] && ri[tid + o] < ri[tid]))) {
                rv[tid] = rv[tid + o]; ri[tid] = ri[tid + o];
            }
            __syncthreads();
        }
        if (!tid) { sel[b * KSEL + s] = ri[0]; sv[ri[0]] = -1e30f; }
        __syncthreads();
    }
}

// ---------------- gather rows (f32 -> bf16) via index table ----------------
__global__ void k_gather_rows_bf16(const float* hidden, const int* idx,
                                   unsigned short* out, int rowsPerB, int T,
                                   int D, long total)
{
    long i = (long)blockIdx.x * blockDim.x + threadIdx.x;
    if (i >= total) return;
    int d = (int)(i % D);
    long rr = i / D;
    int k = (int)(rr % rowsPerB);
    int b = (int)(rr / rowsPerB);
    int r = idx[b * rowsPerB + k];
    out[i] = f2bf(hidden[((long)b * T + r) * D + d]);
}

// ---------------- gather retro rows into memory[:,48:64,:] (f32) ----------------
__global__ void k_gather_retro(const float* hidden, const int* sel, float* mem,
                               int T, int D, int KSEL, long total)
{
    long i = (long)blockIdx.x * blockDim.x + threadIdx.x;
    if (i >= total) return;
    int d = (int)(i % D);
    long rr = i / D;
    int j = (int)(rr % KSEL);
    int b = (int)(rr / KSEL);
    int r = sel[b * KSEL + j];
    mem[((long)b * 64 + 48 + j) * D + d] = hidden[((long)b * T + r) * D + d];
}

// ---------------- context = mean over T (f32 + bf16) ----------------
__global__ void k_context(const float* hidden, float* ctx, unsigned short* ctxb,
                          int T, int D)
{
    int b = blockIdx.x, tid = threadIdx.x;
    for (int d = tid; d < D; d += 256) {
        float s = 0.f;
        for (int t = 0; t < T; ++t) s += hidden[((long)b * T + t) * D + d];
        s /= (float)T;
        ctx[(long)b * D + d] = s;
        ctxb[(long)b * D + d] = f2bf(s);
    }
}

// ---------------- gather hidden[:, T-2, :] (bf16 copy) ----------------
__global__ void k_gather_m2(const unsigned short* hb, unsigned short* out,
                            int T, int D, long total)
{
    long i = (long)blockIdx.x * blockDim.x + threadIdx.x;
    if (i >= total) return;
    int b = (int)(i / D), d = (int)(i % D);
    out[i] = hb[((long)b * T + (T - 2)) * D + d];
}

// ---------------- row softmax (scaled) -> bf16 ----------------
__global__ void k_softmax_rows(const float* S, unsigned short* Ab, int cols,
                               float scale)
{
    int row = blockIdx.x, tid = threadIdx.x;
    const float* sr = S + (long)row * cols;
    __shared__ float red[256];
    float mx = -1e30f;
    for (int c = tid; c < cols; c += 256) mx = fmaxf(mx, sr[c] * scale);
    red[tid] = mx; __syncthreads();
    for (int o = 128; o; o >>= 1) { if (tid < o) red[tid] = fmaxf(red[tid], red[tid + o]); __syncthreads(); }
    mx = red[0]; __syncthreads();
    float sm = 0.f;
    for (int c = tid; c < cols; c += 256) sm += expf(sr[c] * scale - mx);
    red[tid] = sm; __syncthreads();
    for (int o = 128; o; o >>= 1) { if (tid < o) red[tid] += red[tid + o]; __syncthreads(); }
    float inv = 1.f / red[0];
    for (int c = tid; c < cols; c += 256)
        Ab[(long)row * cols + c] = f2bf(expf(sr[c] * scale - mx) * inv);
}

// ---------------- memory attention: scores, softmax, weighted sum -> bf16 ctx ----------------
__global__ void k_mem_attend(const float* mem, const float* qh,
                             unsigned short* ctxb, int D)
{
    int b = blockIdx.x, tid = threadIdx.x;
    int wid = tid >> 5, lane = tid & 31;
    __shared__ float sc[64];
    __shared__ float aw[64];
    for (int m = wid; m < 64; m += 8) {
        const float* mr = mem + ((long)b * 64 + m) * D;
        float s = 0.f;
        for (int d = lane; d < D; d += 32) s += mr[d] * qh[(long)b * D + d];
        for (int o = 16; o; o >>= 1) s += __shfl_down(s, o, 32);
        if (!lane) sc[m] = s;
    }
    __syncthreads();
    if (!tid) {
        float mx = -1e30f;
        for (int m = 0; m < 64; ++m) mx = fmaxf(mx, sc[m]);
        float sm = 0.f;
        for (int m = 0; m < 64; ++m) { aw[m] = expf(sc[m] - mx); sm += aw[m]; }
        float inv = 1.f / sm;
        for (int m = 0; m < 64; ++m) aw[m] *= inv;
    }
    __syncthreads();
    for (int d = tid; d < D; d += 256) {
        float s = 0.f;
        for (int m = 0; m < 64; ++m) s += aw[m] * mem[((long)b * 64 + m) * D + d];
        ctxb[(long)b * D + d] = f2bf(s);
    }
}

// =====================================================================
extern "C" void kernel_launch(void* const* d_in, const int* in_sizes, int n_in,
                              void* d_out, int out_size, void* d_ws, size_t ws_size,
                              hipStream_t stream)
{
    const int V = 50257, D = 512, B = 64, T = 512, FWDN = 48, RETN = 16;
    const int NT = B * T;        // 32768
    const int NC = T - 3;        // 509

    const int*   seq   = (const int*)  d_in[0];
    const float* emb   = (const float*)d_in[1];
    const float* ff_w1 = (const float*)d_in[2];
    const float* ff_b1 = (const float*)d_in[3];
    const float* ff_w2 = (const float*)d_in[4];
    const float* ff_b2 = (const float*)d_in[5];
    const float* ln_g  = (const float*)d_in[6];
    const float* ln_b  = (const float*)d_in[7];
    const float* fg_w  = (const float*)d_in[8];
    const float* fg_b  = (const float*)d_in[9];
    const float* nw_w1 = (const float*)d_in[10];
    const float* nw_b1 = (const float*)d_in[11];
    const float* nw_w2 = (const float*)d_in[12];
    const float* nw_b2 = (const float*)d_in[13];
    const float* wq    = (const float*)d_in[14];
    const float* bq    = (const float*)d_in[15];
    const float* wk    = (const float*)d_in[16];
    const float* bk    = (const float*)d_in[17];
    const float* wv    = (const float*)d_in[18];
    const float* bv    = (const float*)d_in[19];
    const float* wo    = (const float*)d_in[20];
    const float* bo    = (const float*)d_in[21];
    const float* rq_w  = (const float*)d_in[22];
    const float* rq_b  = (const float*)d_in[23];
    const float* out_w = (const float*)d_in[24];
    const float* out_b = (const float*)d_in[25];
    float* dout = (float*)d_out;

    // ---------------- workspace layout (reused regions) ----------------
    char* ws = (char*)d_ws;
    float*          hid  = (float*)(ws);                       // 64 MiB: h0 -> x -> hidden
    unsigned short* hb16 = (unsigned short*)(ws + 67108864);   // 32 MiB: h0 bf16 -> hidden bf16
    unsigned short* a1   = (unsigned short*)(ws + 100663296);  // 64 MiB: FF hidden bf16
    unsigned short* kkb  = a1;                                 // reuse: K bf16 (32 MiB)
    unsigned short* vb   = (unsigned short*)(ws + 134217728);  // reuse: V bf16 (32 MiB)
    float*          gsh  = (float*)(ws + 100663296);           // reuse late: gate hidden f32
    size_t so = 167772160;                                     // small region @160 MiB
    auto salloc = [&](size_t bytes) { char* p = ws + so; so = (so + bytes + 255) & ~(size_t)255; return p; };
    float*          fsc    = (float*)salloc((size_t)B * NC * 4);
    int*            fidx   = (int*)  salloc((size_t)B * FWDN * 4);
    int*            sel    = (int*)  salloc((size_t)B * RETN * 4);
    unsigned short* fwdh   = (unsigned short*)salloc((size_t)B * FWDN * D * 2);
    unsigned short* qb     = (unsigned short*)salloc((size_t)B * FWDN * D * 2);
    float*          scoresF= (float*)salloc((size_t)B * FWDN * T * 4);
    unsigned short* attnb  = (unsigned short*)salloc((size_t)B * FWDN * T * 2);
    unsigned short* reb    = (unsigned short*)salloc((size_t)B * FWDN * D * 2);
    float*          memf   = (float*)salloc((size_t)B * 64 * D * 4);
    float*          ctxterm= (float*)salloc((size_t)B * D * 4);
    float*          ctxf   = (float*)salloc((size_t)B * D * 4);
    unsigned short* ctxb16 = (unsigned short*)salloc((size_t)B * D * 2);
    float*          gsbuf  = (float*)salloc((size_t)B * NC * 4);
    unsigned short* hm2b   = (unsigned short*)salloc((size_t)B * D * 2);
    float*          qhf    = (float*)salloc((size_t)B * D * 4);
    unsigned short* ctx2b  = (unsigned short*)salloc((size_t)B * D * 2);
    // pre-converted bf16 weights (one-time per launch)
    unsigned short* ff_w1b = (unsigned short*)salloc((size_t)D * 2 * D * 2);
    unsigned short* ff_w2b = (unsigned short*)salloc((size_t)2 * D * D * 2);
    unsigned short* nw_w1b = (unsigned short*)salloc((size_t)2 * D * D * 2);
    unsigned short* wqb    = (unsigned short*)salloc((size_t)D * D * 2);
    unsigned short* wkb    = (unsigned short*)salloc((size_t)D * D * 2);
    unsigned short* wvb    = (unsigned short*)salloc((size_t)D * D * 2);
    unsigned short* wob    = (unsigned short*)salloc((size_t)D * D * 2);
    unsigned short* rq_wb  = (unsigned short*)salloc((size_t)D * D * 2);

    dim3 blk(256);
    auto cdiv = [](long a, long b) { return (int)((a + b - 1) / b); };
    auto G = [&](const unsigned short* A, long lda, long ab,
                 const void* Bm, long ldb, long bb, int bT, int bBf,
                 float* C, long ldc, long cb,
                 unsigned short* Cb, long cbb,
                 const float* bias, long biasb,
                 const float* res, long rb,
                 int M, int N, int K, int nb, int relu, int skipF)
    {
        int flags = (relu ? F_RELU : 0) | (skipF ? F_SKIPF32 : 0) | (Cb ? F_BF16OUT : 0);
        dim3 g(cdiv(N, 64), cdiv(M, 64), nb);
        if (bBf && bT)
            k_gemm<1, 1><<<g, blk, 0, stream>>>(A, lda, ab, Bm, ldb, bb,
                C, ldc, cb, Cb, cbb, bias, biasb, res, rb, M, N, K, flags);
        else if (bBf)
            k_gemm<1, 0><<<g, blk, 0, stream>>>(A, lda, ab, Bm, ldb, bb,
                C, ldc, cb, Cb, cbb, bias, biasb, res, rb, M, N, K, flags);
        else
            k_gemm<0, 0><<<g, blk, 0, stream>>>(A, lda, ab, Bm, ldb, bb,
                C, ldc, cb, Cb, cbb, bias, biasb, res, rb, M, N, K, flags);
    };
    auto CVT = [&](const float* in, unsigned short* out, long n) {
        k_cvt_bf16<<<cdiv(n, 256), blk, 0, stream>>>(in, out, n);
    };

    // 0) one-time weight conversions to bf16 (~5.8 MB total)
    CVT(ff_w1, ff_w1b, (long)D * 2 * D);
    CVT(ff_w2, ff_w2b, (long)2 * D * D);
    CVT(nw_w1, nw_w1b, (long)2 * D * D);
    CVT(wq, wqb, (long)D * D);
    CVT(wk, wkb, (long)D * D);
    CVT(wv, wvb, (long)D * D);
    CVT(wo, wob, (long)D * D);
    CVT(rq_w, rq_wb, (long)D * D);

    // 1) embed gather
    {
        long tot = (long)NT * D;
        k_embed<<<cdiv(tot, 256), blk, 0, stream>>>(seq, emb, hid, hb16, tot, D);
    }
    // 2) FF layer 1: relu(h0 @ ff_w1 + b1) -> a1 (bf16 only)
    G(hb16, D, 0, ff_w1b, 2 * D, 0, 0, 1, nullptr, 2 * D, 0, a1, 0,
      ff_b1, 0, nullptr, 0, NT, 2 * D, D, 1, 1, 1);
    // 3) FF layer 2 + residual: x = a1 @ ff_w2 + b2 + h0 -> hid (in place)
    G(a1, 2 * D, 0, ff_w2b, D, 0, 0, 1, hid, D, 0, nullptr, 0,
      ff_b2, 0, hid, 0, NT, D, 2 * D, 1, 0, 0);
    // 4) LayerNorm in place -> hidden (f32 + bf16)
    k_ln<<<NT, blk, 0, stream>>>(hid, hb16, ln_g, ln_b, D);
    // 5) K = hidden @ wk + bk (bf16 only)
    G(hb16, D, 0, wkb, D, 0, 0, 1, nullptr, D, 0, kkb, 0,
      bk, 0, nullptr, 0, NT, D, D, 1, 0, 1);
    // 6) V = hidden @ wv + bv (bf16 only)
    G(hb16, D, 0, wvb, D, 0, 0, 1, nullptr, D, 0, vb, 0,
      bv, 0, nullptr, 0, NT, D, D, 1, 0, 1);
    // 7) forward scores = cand . fg_w + fg_b
    k_rowdot<<<cdiv((long)B * NC * 32, 256), blk, 0, stream>>>(
        hid, D, (long)T * D, NC, B, fg_w, fg_b, D, 0, fsc, NC);
    // 8) top-48 (sorted ascending)
    k_topk_fwd<<<B, blk, 0, stream>>>(fsc, fidx, NC, FWDN);
    // 9) gather fwd_h -> bf16
    {
        long tot = (long)B * FWDN * D;
        k_gather_rows_bf16<<<cdiv(tot, 256), blk, 0, stream>>>(hid, fidx, fwdh, FWDN, T, D, tot);
    }
    // 10) q = fwd_h @ wq + bq (bf16 only)
    G(fwdh, D, 0, wqb, D, 0, 0, 1, nullptr, D, 0, qb, 0,
      bq, 0, nullptr, 0, B * FWDN, D, D, 1, 0, 1);
    // 11) scores = q @ K^T (batched, B transposed bf16)
    G(qb, D, (long)FWDN * D, kkb, D, (long)T * D, 1, 1,
      scoresF, T, (long)FWDN * T, nullptr, 0, nullptr, 0, nullptr, 0,
      FWDN, T, D, B, 0, 0);
    // 12) softmax(scores / sqrt(D)) -> attn bf16
    k_softmax_rows<<<B * FWDN, blk, 0, stream>>>(scoresF, attnb, T, 1.0f / sqrtf((float)D));
    // 13) re = attn @ V (batched, bf16 B)
    G(attnb, T, (long)FWDN * T, vb, D, (long)T * D, 0, 1,
      nullptr, D, 0, reb, (long)FWDN * D, nullptr, 0, nullptr, 0,
      FWDN, D, T, B, 0, 1);
    // 14) re_slots = re @ wo + bo -> memory rows [0,48)
    G(reb, D, (long)FWDN * D, wob, D, 0, 0, 1,
      memf, D, (long)64 * D, nullptr, 0, bo, 0, nullptr, 0,
      FWDN, D, D, B, 0, 0);
    // 15) context = mean_t(hidden)
    k_context<<<B, blk, 0, stream>>>(hid, ctxf, ctxb16, T, D);
    // 16) ctxterm = context @ nw_w1[D:2D,:] + nw_b1
    G(ctxb16, D, 0, nw_w1b + (long)D * D, D, 0, 0, 1,
      ctxterm, D, 0, nullptr, 0, nw_b1, 0, nullptr, 0, B, D, D, 1, 0, 0);
    // 17) gate hidden = relu(cand @ nw_w1[:D,:] + ctxterm[b]) (batched bias)
    G(hb16, D, (long)T * D, nw_w1b, D, 0, 0, 1,
      gsh, D, (long)NC * D, nullptr, 0, ctxterm, D, nullptr, 0,
      NC, D, D, B, 1, 0);
    // 18) gs = sigmoid(gate_hidden . nw_w2 + nw_b2)
    k_rowdot<<<cdiv((long)B * NC * 32, 256), blk, 0, stream>>>(
        gsh, D, (long)NC * D, NC, B, nw_w2, nw_b2, D, 1, gsbuf, NC);
    // 19) retro top-16 (fwd picks masked, descending order)
    k_topk_retro<<<B, blk, 0, stream>>>(gsbuf, fidx, sel, NC, FWDN, RETN);
    // 20) gather retro rows -> memory rows [48,64)
    {
        long tot = (long)B * RETN * D;
        k_gather_retro<<<cdiv(tot, 256), blk, 0, stream>>>(hid, sel, memf, T, D, RETN, tot);
    }
    // 21) gather hidden[:, T-2, :] (bf16)
    {
        long tot = (long)B * D;
        k_gather_m2<<<cdiv(tot, 256), blk, 0, stream>>>(hb16, hm2b, T, D, tot);
    }
    // 22) qh = hm2 @ rq_w + rq_b
    G(hm2b, D, 0, rq_wb, D, 0, 0, 1, qhf, D, 0, nullptr, 0,
      rq_b, 0, nullptr, 0, B, D, D, 1, 0, 0);
    // 23) memory attention -> ctx (bf16)
    k_mem_attend<<<B, blk, 0, stream>>>(memf, qhf, ctx2b, D);
    // 24) out = ctx @ out_w + out_b
    G(ctx2b, D, 0, out_w, V, 0, 0, 0, dout, V, 0, nullptr, 0,
      out_b, 0, nullptr, 0, B, V, D, 1, 0, 0);
}
